// RootAlign_40570261078532
// MI455X (gfx1250) — matrix operand order repaired and checked by reading, not compiled, not run on hardware
//
#include <hip/hip_runtime.h>
#include <hip/hip_bf16.h>
#include <math.h>

typedef __attribute__((ext_vector_type(16))) _Float16 v16h;
typedef __attribute__((ext_vector_type(8)))  _Float16 v8h;
typedef __attribute__((ext_vector_type(8)))  float    v8f;

#define WD        300     // word dim
#define HID       300     // hidden
#define RELS      3
#define BATCH     512
#define LEAVES    256

#define NT        19      // N tiles of 16 -> 304 (covers 300)
#define KC_LEAF   10      // leaf K chunks of 32 -> 320 (covers 300)
#define KC_COMB   20      // combine K chunks of 32 -> 640 (covers 2*320 row-pair)
#define STRIDE    320     // LDS row stride in f16 elements
#define FRAGH     512     // halves per packed B fragment (32 lanes * 16)

#define WLEAF_HALVES (NT * KC_LEAF * FRAGH)   // 97280
#define WH_HALVES    (NT * KC_COMB * FRAGH)   // 194560

// ---------------- weight pre-pack kernels ----------------
// B-fragment layout (v_wmma_f32_16x16x32_f16): within a 512-half fragment,
// lane L holds N column (16*nt + L%16), element e holds K (32c + (e&7) + 8*(L/16) + 16*(e>>3)).

__global__ void pack_wleaf(const float* __restrict__ wl, _Float16* __restrict__ dst) {
    int idx = blockIdx.x * 256 + threadIdx.x;
    if (idx >= WLEAF_HALVES) return;
    int frag = idx >> 9;
    int r    = idx & 511;
    int lane = r >> 4;
    int e    = r & 15;
    int nt = frag / KC_LEAF, ck = frag % KC_LEAF;
    int k = ck * 32 + (e & 7) + ((lane >> 4) << 3) + ((e >> 3) << 4);          // input feature
    int n = nt * 16 + (lane & 15);             // output feature
    float v = (k < WD && n < HID) ? wl[n * WD + k] : 0.0f;  // B[k][n] = W_leaf[n][k]
    dst[idx] = (_Float16)v;
}

__global__ void pack_wh(const float* __restrict__ wh, _Float16* __restrict__ dst) {
    int idx = blockIdx.x * 256 + threadIdx.x;
    if (idx >= WH_HALVES) return;
    int frag = idx >> 9;
    int r    = idx & 511;
    int lane = r >> 4;
    int e    = r & 15;
    int nt = frag / KC_COMB, ck = frag % KC_COMB;
    int kp = ck * 32 + (e & 7) + ((lane >> 4) << 3) + ((e >> 3) << 4);         // 0..639 over the 2*STRIDE row pair
    int n  = nt * 16 + (lane & 15);
    float v = 0.0f;
    if (n < HID) {
        if (kp < 300)                     v = wh[n * 600 + kp];         // left child
        else if (kp >= 320 && kp < 620)   v = wh[n * 600 + (kp - 20)];  // right child (kp-320 -> col 300+..)
    }
    dst[idx] = (_Float16)v;
}

__global__ void pack_bias(const float* __restrict__ bh, float* __restrict__ dst) {
    int n = threadIdx.x;
    if (n < NT * 16) dst[n] = (n < HID) ? bh[n] : 0.0f;
}

// ---------------- fused tree kernel ----------------
// A-fragment (16x32 f16) per ISA: lane L<16 -> row M=L, elems 0..7 = K 0..7,
// elems 8..15 = K 16..23; lane L>=16 -> same row, K offset +8.
__device__ __forceinline__ v16h load_A(const _Float16* rowp, int lane) {
    v16h a;
    const _Float16* p = rowp + (((lane >> 4) & 1) << 3);
    ((v8h*)&a)[0] = *(const v8h*)(p);        // K block lo
    ((v8h*)&a)[1] = *(const v8h*)(p + 16);   // K block hi
    return a;
}

__device__ __forceinline__ v16h load_B(const _Float16* pack, int frag, int lane) {
    v16h b;
    const _Float16* p = pack + (size_t)frag * FRAGH + lane * 16;
    ((v8h*)&b)[0] = *(const v8h*)(p);
    ((v8h*)&b)[1] = *(const v8h*)(p + 8);
    return b;
}

__global__ void __launch_bounds__(256)
tree_kernel(const int*   __restrict__ word_ids,
            const float* __restrict__ emb,
            const _Float16* __restrict__ wleaf_pack,
            const _Float16* __restrict__ wh_pack,
            const float* __restrict__ bias_pad,
            const float* __restrict__ w_cls,
            const float* __restrict__ b_cls,
            float* __restrict__ out)
{
    extern __shared__ __align__(16) char smem_raw[];
    _Float16* bufA  = (_Float16*)smem_raw;            // 256 x 320 f16
    _Float16* bufX  = bufA + 256 * STRIDE;            // 128 x 320 f16
    float*    rootf = (float*)(bufX + 128 * STRIDE);  // 2 x 304 f32 root stash

    const int b    = blockIdx.x;
    const int tid  = threadIdx.x;
    const int wave = tid >> 5;
    const int lane = tid & 31;

    for (int s = 0; s < 2; ++s) {
        // ================= leaf phase (two halves of 128 leaves) =================
        for (int h = 0; h < 2; ++h) {
            __syncthreads();
            // gather 128 embedding rows (f32 -> f16) into bufX, pad cols >= 300
            {
                int g    = tid >> 1;                  // row 0..127
                int q    = tid & 1;
                int leaf = h * 128 + g;
                int wid  = word_ids[((size_t)b * 2 + s) * LEAVES + leaf];
                const float* er = emb + (size_t)wid * WD;
                for (int c = q; c < STRIDE; c += 2) {
                    float v = (c < WD) ? er[c] : 0.0f;
                    bufX[g * STRIDE + c] = (_Float16)v;
                }
            }
            __syncthreads();
            // GEMM (128 x 320) x packed W_leaf -> bufA rows [h*128, h*128+128)
            const int T = 8 * NT; // 8 M-tiles * 19 N-tiles
            for (int idx = wave; idx < T; idx += 8) {
                int mt = idx / NT;
                int nt = idx % NT;
                v8f acc = {};
                int arow = mt * 16 + (lane & 15);
                for (int ck = 0; ck < KC_LEAF; ++ck) {
                    v16h a = load_A(bufX + arow * STRIDE + ck * 32, lane);
                    v16h bf = load_B(wleaf_pack, nt * KC_LEAF + ck, lane);
                    acc = __builtin_amdgcn_wmma_f32_16x16x32_f16(
                        false, a, false, bf, (short)0, acc, false, false);
                }
                int ncol  = nt * 16 + (lane & 15);
                int rbase = h * 128 + mt * 16 + (((lane >> 4) & 1) << 3);
                #pragma unroll
                for (int r = 0; r < 8; ++r)
                    bufA[(rbase + r) * STRIDE + ncol] = (_Float16)acc[r];
            }
        }

        // ================= combine levels (ping-pong in LDS) =================
        _Float16* cur = bufA;
        _Float16* nxt = bufX;
        int nIn = 256;
        for (int level = 0; level < 8; ++level) {
            __syncthreads();
            int Mout   = nIn >> 1;
            int Mtiles = (Mout + 15) >> 4;
            int T      = Mtiles * NT;
            for (int idx = wave; idx < T; idx += 8) {
                int mt = idx / NT;
                int nt = idx % NT;
                v8f acc = {};
                int m = mt * 16 + (lane & 15);
                for (int ck = 0; ck < KC_COMB; ++ck) {
                    v16h a = {};
                    if (m < Mout)  // rows 2m,2m+1 are contiguous (640 halves)
                        a = load_A(cur + (size_t)m * (2 * STRIDE) + ck * 32, lane);
                    v16h bf = load_B(wh_pack, nt * KC_COMB + ck, lane);
                    acc = __builtin_amdgcn_wmma_f32_16x16x32_f16(
                        false, a, false, bf, (short)0, acc, false, false);
                }
                int   ncol = nt * 16 + (lane & 15);
                float bias = bias_pad[ncol];
                if (Mout == 1) {
                    // root row (M=0) lives in element 0 of lanes 0..15; keep f32
                    if (lane < 16) rootf[s * (NT * 16) + ncol] = acc[0] + bias;
                } else {
                    int rbase = mt * 16 + (((lane >> 4) & 1) << 3);
                    #pragma unroll
                    for (int r = 0; r < 8; ++r) {
                        int mrow = rbase + r;
                        if (mrow < Mout)
                            nxt[mrow * STRIDE + ncol] = (_Float16)(acc[r] + bias);
                    }
                }
            }
            _Float16* t = cur; cur = nxt; nxt = t;
            nIn = Mout;
        }
    }

    __syncthreads();
    // ================= classifier: sigmoid -> 3 logits -> log_softmax =================
    if (tid < 32) {
        float logits[RELS];
        for (int r2 = 0; r2 < RELS; ++r2) {
            float part = 0.0f;
            for (int j = lane; j < 2 * HID; j += 32) {
                int s2 = j / HID, n2 = j % HID;
                float x = rootf[s2 * (NT * 16) + n2];
                float f = 1.0f / (1.0f + __expf(-x));
                part += f * w_cls[r2 * (2 * HID) + j];
            }
            #pragma unroll
            for (int m2 = 16; m2 >= 1; m2 >>= 1)
                part += __shfl_xor(part, m2, 32);
            logits[r2] = part + b_cls[r2];
        }
        if (lane == 0) {
            float mx = fmaxf(logits[0], fmaxf(logits[1], logits[2]));
            float sum = 0.0f;
            #pragma unroll
            for (int r2 = 0; r2 < RELS; ++r2) sum += __expf(logits[r2] - mx);
            float lse = mx + __logf(sum);
            #pragma unroll
            for (int r2 = 0; r2 < RELS; ++r2)
                out[b * RELS + r2] = logits[r2] - lse;
        }
    }
}

// ---------------- host launch ----------------
extern "C" void kernel_launch(void* const* d_in, const int* in_sizes, int n_in,
                              void* d_out, int out_size, void* d_ws, size_t ws_size,
                              hipStream_t stream) {
    (void)in_sizes; (void)n_in; (void)out_size; (void)ws_size;
    const int*   word_ids = (const int*)  d_in[0];
    const float* emb      = (const float*)d_in[1];
    const float* wleaf    = (const float*)d_in[2];
    const float* wh       = (const float*)d_in[3];
    const float* bh       = (const float*)d_in[4];
    const float* wcls     = (const float*)d_in[5];
    const float* bcls     = (const float*)d_in[6];

    _Float16* wleaf_pack = (_Float16*)d_ws;
    _Float16* wh_pack    = wleaf_pack + WLEAF_HALVES;
    float*    bias_pad   = (float*)((char*)d_ws + (size_t)(WLEAF_HALVES + WH_HALVES) * 2);

    pack_wleaf<<<(WLEAF_HALVES + 255) / 256, 256, 0, stream>>>(wleaf, wleaf_pack);
    pack_wh<<<(WH_HALVES + 255) / 256, 256, 0, stream>>>(wh, wh_pack);
    pack_bias<<<1, 512, 0, stream>>>(bh, bias_pad);

    size_t smem = (size_t)(256 * STRIDE + 128 * STRIDE) * sizeof(_Float16)
                + (size_t)2 * (NT * 16) * sizeof(float);   // 248192 B (<320KB/WG)
    (void)hipFuncSetAttribute(reinterpret_cast<const void*>(tree_kernel),
                              hipFuncAttributeMaxDynamicSharedMemorySize, (int)smem);
    tree_kernel<<<BATCH, 256, smem, stream>>>(word_ids, emb, wleaf_pack, wh_pack,
                                              bias_pad, wcls, bcls, (float*)d_out);
}
